// GATAndMLP_13967233647439
// MI455X (gfx1250) — matrix operand order repaired and checked by reading, not compiled, not run on hardware
//
#include <hip/hip_runtime.h>
#include <hip/hip_bf16.h>

#define NN      50000
#define NE      800000
#define EMBED   128
#define HEADS   8
#define DH      16
#define HIDDEN  512
#define OUTD    128
#define EPSV    1e-5f
#define NSLOPE  0.2f
#define LDSK    136   // padded K stride for LDS B tiles (16B aligned, bank-skewed)

typedef __attribute__((ext_vector_type(16))) __bf16 v16bf;
typedef __attribute__((ext_vector_type(8)))  __bf16 v8bf;
typedef __attribute__((ext_vector_type(8)))  float  v8f;

#if defined(__has_builtin)
#if __has_builtin(__builtin_amdgcn_global_load_async_to_lds_b128) && \
    __has_builtin(__builtin_amdgcn_s_wait_asynccnt)
#define USE_ASYNC_LDS 1
#endif
#endif

#if defined(USE_ASYNC_LDS)
typedef int v4i_b __attribute__((vector_size(16)));
typedef __attribute__((address_space(1))) v4i_b* gvec128_t;
typedef __attribute__((address_space(3))) v4i_b* lvec128_t;
#endif

static __device__ __forceinline__ void atomicAddF(float* p, float v) {
#if defined(__HIP_PLATFORM_AMD__)
  unsafeAtomicAdd(p, v);   // native global_atomic_add_f32 (L2 atomic units)
#else
  atomicAdd(p, v);
#endif
}

// ---------------- WMMA helpers ----------------

static __device__ __forceinline__ v8f wmma_bf16(v16bf a, v16bf b, v8f c) {
  // (neg_a, A, neg_b, B, c_mod, C, reuse_a, reuse_b)
  return __builtin_amdgcn_wmma_f32_16x16x32_bf16(false, a, false, b, (short)0, c,
                                                 false, false);
}

// A (16x32 bf16) lane layout: row M = lane%16; lanes 0-15 hold K=[0,8)+[16,24),
// lanes 16-31 hold K=[8,16)+[24,32). Source is f32 row-major; convert on load.
static __device__ __forceinline__ v16bf load_A_f32(const float* __restrict__ A, int row,
                                                   int ldk, int kbase, int lane,
                                                   bool valid, const float* __restrict__ bias) {
  int koff = kbase + ((lane >> 4) << 3);
  float f[16];
  if (valid) {
    const float* p = A + (size_t)row * ldk + koff;
    float4 a0 = *(const float4*)(p);
    float4 a1 = *(const float4*)(p + 4);
    float4 b0 = *(const float4*)(p + 16);
    float4 b1 = *(const float4*)(p + 20);
    f[0] = a0.x; f[1] = a0.y; f[2]  = a0.z; f[3]  = a0.w;
    f[4] = a1.x; f[5] = a1.y; f[6]  = a1.z; f[7]  = a1.w;
    f[8] = b0.x; f[9] = b0.y; f[10] = b0.z; f[11] = b0.w;
    f[12] = b1.x; f[13] = b1.y; f[14] = b1.z; f[15] = b1.w;
  } else {
    #pragma unroll
    for (int i = 0; i < 16; ++i) f[i] = 0.f;
  }
  if (bias) {
    #pragma unroll
    for (int i = 0; i < 8; ++i) { f[i] += bias[koff + i]; f[i + 8] += bias[koff + 16 + i]; }
  }
  v16bf a;
  #pragma unroll
  for (int i = 0; i < 16; ++i) a[i] = (__bf16)f[i];
  return a;
}

static __device__ __forceinline__ v16bf load_A_bf16(const __bf16* __restrict__ A, int row,
                                                    int ldk, int kbase, int lane, bool valid) {
  v16bf a;
  int koff = kbase + ((lane >> 4) << 3);
  if (valid) {
    const v8bf* p0 = (const v8bf*)(A + (size_t)row * ldk + koff);
    const v8bf* p1 = (const v8bf*)(A + (size_t)row * ldk + koff + 16);
    v8bf lo = *p0, hi = *p1;
    #pragma unroll
    for (int i = 0; i < 8; ++i) { a[i] = lo[i]; a[i + 8] = hi[i]; }
  } else {
    #pragma unroll
    for (int i = 0; i < 16; ++i) a[i] = (__bf16)0.f;
  }
  return a;
}

// B (32x16 bf16) lane layout: column N = lane%16; lanes 0-15 hold K=[0,8)+[16,24),
// lanes 16-31 hold K=[8,16)+[24,32) (same as A). Wt in LDS is stored N-major
// (Wt[n][k]) so a column is contiguous -> two ds_load_b128 per operand.
static __device__ __forceinline__ v16bf load_B_lds(const __bf16* Wt, int nt, int kbase,
                                                   int lane) {
  int n = nt * 16 + (lane & 15);
  int koff = kbase + ((lane >> 4) << 3);
  const v8bf* p = (const v8bf*)(Wt + n * LDSK + koff);
  v8bf lo = p[0], hi = p[2];
  v16bf b;
  #pragma unroll
  for (int i = 0; i < 8; ++i) { b[i] = lo[i]; b[i + 8] = hi[i]; }
  return b;
}

// Stage a 128x128 bf16 tile (row stride g_ld elements) into LDS Wt[n*LDSK+k].
// Uses ASYNCcnt-tracked global_load_async_to_lds_b128 when the builtin exists.
static __device__ __forceinline__ void stage_tile(const __bf16* __restrict__ gsrc, int g_ld,
                                                  __bf16* lds, int t) {
#if defined(USE_ASYNC_LDS)
  #pragma unroll
  for (int i = t; i < 2048; i += 256) {     // 2048 packets of 16B (8 bf16)
    int n = i >> 4;
    int kp = (i & 15) << 3;
    __builtin_amdgcn_global_load_async_to_lds_b128(
        (gvec128_t)(gsrc + (size_t)n * g_ld + kp),
        (lvec128_t)(lds + n * LDSK + kp), 0, 0);
  }
  __builtin_amdgcn_s_wait_asynccnt(0);
#else
  for (int i = t; i < 2048; i += 256) {
    int n = i >> 4;
    int kp = (i & 15) << 3;
    *(v8bf*)(lds + n * LDSK + kp) = *(const v8bf*)(gsrc + (size_t)n * g_ld + kp);
  }
#endif
}

// C/D f32 layout: VGPR r, lanes 0-15 -> (M=r, N=lane); lanes 16-31 -> (M=8+r, N=lane-16)
static __device__ __forceinline__ void store_C_f32(float* __restrict__ out, int ldn,
                                                   int mbase, int nbase, int lane, v8f c,
                                                   int mlimit, float badd) {
  int m0 = mbase + ((lane >> 4) << 3);
  int n  = nbase + (lane & 15);
  if (mbase + 16 <= mlimit) {               // wave-uniform fast path (3125/3128 tiles)
    #pragma unroll
    for (int r = 0; r < 8; ++r) out[(size_t)(m0 + r) * ldn + n] = c[r] + badd;
  } else {
    #pragma unroll
    for (int r = 0; r < 8; ++r) {
      int row = m0 + r;
      if (row < mlimit) out[(size_t)row * ldn + n] = c[r] + badd;
    }
  }
}

// ---------------- weight prep: f32 -> transposed bf16 ----------------

__global__ void k_prep_weights(const float* __restrict__ Wfc, const float* __restrict__ W1,
                               const float* __restrict__ W2,
                               unsigned short* __restrict__ wtfc_u,
                               unsigned short* __restrict__ wt1_u,
                               unsigned short* __restrict__ wt2_u) {
  __bf16* wtfc = (__bf16*)wtfc_u;
  __bf16* wt1  = (__bf16*)wt1_u;
  __bf16* wt2  = (__bf16*)wt2_u;
  int gid = blockIdx.x * blockDim.x + threadIdx.x;
  if (gid < EMBED * EMBED) {                      // Wt_fc[n][k] = W_fc[k][n]
    int n = gid >> 7, k = gid & 127;
    wtfc[gid] = (__bf16)Wfc[(size_t)k * EMBED + n];
  } else if (gid < EMBED * EMBED + HIDDEN * EMBED) {   // Wt1[n][k] = W1[k][n]
    int i = gid - EMBED * EMBED;
    int n = i >> 7, k = i & 127;
    wt1[i] = (__bf16)W1[(size_t)k * HIDDEN + n];
  } else if (gid < EMBED * EMBED + 2 * HIDDEN * EMBED) {  // Wt2[n][k] = W2[k][n]
    int i = gid - EMBED * EMBED - HIDDEN * EMBED;
    int n = i >> 9, k = i & 511;
    wt2[i] = (__bf16)W2[(size_t)k * OUTD + n];
  }
}

// ---------------- GEMM kernels ----------------

// feat = h @ W_fc  [50000x128]@[128x128]
__global__ void __launch_bounds__(256) k_gemm_feat(const float* __restrict__ h,
                                                   const unsigned short* __restrict__ wtfc_u,
                                                   float* __restrict__ feat) {
  __shared__ __bf16 Wt[128 * LDSK];
  int t = threadIdx.x;
  stage_tile((const __bf16*)wtfc_u, 128, Wt, t);
  __syncthreads();
  int wave = t >> 5, lane = t & 31;
  int mbase = blockIdx.x * 128 + wave * 16;
  int row = mbase + (lane & 15);
  bool valid = row < NN;
  v16bf a[4];
  #pragma unroll
  for (int kc = 0; kc < 4; ++kc)
    a[kc] = load_A_f32(h, row, EMBED, kc * 32, lane, valid, nullptr);
  #pragma unroll
  for (int nt = 0; nt < 8; ++nt) {
    v8f c = {};
    #pragma unroll
    for (int kc = 0; kc < 4; ++kc)
      c = wmma_bf16(a[kc], load_B_lds(Wt, nt, kc * 32, lane), c);
    store_C_f32(feat, EMBED, mbase, nt * 16, lane, c, NN, 0.f);
  }
}

// x1 = relu((rst + gat_bias) @ W1 + b1) -> bf16   [50000x128]@[128x512]
__global__ void __launch_bounds__(256) k_gemm_mlp1(const float* __restrict__ rst,
                                                   const float* __restrict__ gbias,
                                                   const unsigned short* __restrict__ wt1_u,
                                                   const float* __restrict__ b1,
                                                   unsigned short* __restrict__ x1u) {
  __bf16* x1 = (__bf16*)x1u;
  __shared__ __bf16 Wt[128 * LDSK];
  int t = threadIdx.x;
  int ntbase = blockIdx.y * 128;
  stage_tile((const __bf16*)wt1_u + (size_t)ntbase * 128, 128, Wt, t);
  __syncthreads();
  int wave = t >> 5, lane = t & 31;
  int mbase = blockIdx.x * 128 + wave * 16;
  int row = mbase + (lane & 15);
  bool valid = row < NN;
  bool fullm = (mbase + 16 <= NN);
  v16bf a[4];
  #pragma unroll
  for (int kc = 0; kc < 4; ++kc)
    a[kc] = load_A_f32(rst, row, EMBED, kc * 32, lane, valid, gbias);
  #pragma unroll
  for (int nt = 0; nt < 8; ++nt) {
    v8f c = {};
    #pragma unroll
    for (int kc = 0; kc < 4; ++kc)
      c = wmma_bf16(a[kc], load_B_lds(Wt, nt, kc * 32, lane), c);
    int m0 = mbase + ((lane >> 4) << 3);
    int n  = ntbase + nt * 16 + (lane & 15);
    float bb = b1[n];
    if (fullm) {
      #pragma unroll
      for (int r = 0; r < 8; ++r) {
        float v = c[r] + bb;
        v = v > 0.f ? v : 0.f;
        x1[(size_t)(m0 + r) * HIDDEN + n] = (__bf16)v;
      }
    } else {
      #pragma unroll
      for (int r = 0; r < 8; ++r) {
        int rrow = m0 + r;
        if (rrow < NN) {
          float v = c[r] + bb;
          v = v > 0.f ? v : 0.f;
          x1[(size_t)rrow * HIDDEN + n] = (__bf16)v;
        }
      }
    }
  }
}

// y = x1 @ W2 + b2   [50000x512]@[512x128], K-staged through LDS
__global__ void __launch_bounds__(256) k_gemm_mlp2(const unsigned short* __restrict__ x1u,
                                                   const unsigned short* __restrict__ wt2_u,
                                                   const float* __restrict__ b2,
                                                   float* __restrict__ y) {
  const __bf16* x1  = (const __bf16*)x1u;
  const __bf16* wt2 = (const __bf16*)wt2_u;
  __shared__ __bf16 Wt[128 * LDSK];
  int t = threadIdx.x;
  int wave = t >> 5, lane = t & 31;
  int mbase = blockIdx.x * 128 + wave * 16;
  int row = mbase + (lane & 15);
  bool valid = row < NN;
  v8f acc[8];
  #pragma unroll
  for (int nt = 0; nt < 8; ++nt) acc[nt] = (v8f){};
  for (int ks = 0; ks < 4; ++ks) {
    __syncthreads();
    stage_tile(wt2 + ks * 128, HIDDEN, Wt, t);   // Wt2 rows have stride 512
    __syncthreads();
    v16bf a[4];
    #pragma unroll
    for (int kc = 0; kc < 4; ++kc)
      a[kc] = load_A_bf16(x1, row, HIDDEN, ks * 128 + kc * 32, lane, valid);
    #pragma unroll
    for (int nt = 0; nt < 8; ++nt) {
      #pragma unroll
      for (int kc = 0; kc < 4; ++kc)
        acc[nt] = wmma_bf16(a[kc], load_B_lds(Wt, nt, kc * 32, lane), acc[nt]);
    }
  }
  #pragma unroll
  for (int nt = 0; nt < 8; ++nt) {
    int n = nt * 16 + (lane & 15);
    store_C_f32(y, OUTD, mbase, nt * 16, lane, acc[nt], NN, b2[n]);
  }
}

// ---------------- GAT edge-phase kernels ----------------

// order-preserving f32 <-> u32 map so segment-max can use integer atomicMax
static __device__ __forceinline__ unsigned fkey(float f) {
  unsigned u = __float_as_uint(f);
  return (u & 0x80000000u) ? ~u : (u | 0x80000000u);
}
static __device__ __forceinline__ float funkey(unsigned u) {
  unsigned b = (u & 0x80000000u) ? (u & 0x7FFFFFFFu) : ~u;
  return __uint_as_float(b);
}

__global__ void k_init(float* __restrict__ rst, float* __restrict__ denom,
                       unsigned* __restrict__ segkey, float* __restrict__ stats) {
  int gid = blockIdx.x * blockDim.x + threadIdx.x;
  if (gid < NN * EMBED) rst[gid] = 0.f;
  if (gid < NN * HEADS) { denom[gid] = 0.f; segkey[gid] = 0u; }
  if (gid < 256) stats[gid] = 0.f;
}

__global__ void k_el_er(const float* __restrict__ feat, const float* __restrict__ al,
                        const float* __restrict__ ar, float* __restrict__ el,
                        float* __restrict__ er) {
  int gid = blockIdx.x * blockDim.x + threadIdx.x;
  if (gid >= NN * HEADS) return;
  int n = gid >> 3, hh = gid & 7;
  const float* f = feat + (size_t)n * EMBED + hh * DH;
  float sl = 0.f, sr = 0.f;
  #pragma unroll
  for (int d = 0; d < DH; ++d) {
    float v = f[d];
    sl += v * al[hh * DH + d];
    sr += v * ar[hh * DH + d];
  }
  el[gid] = sl;
  er[gid] = sr;
}

__global__ void k_edge_max(const int* __restrict__ src, const int* __restrict__ dst,
                           const float* __restrict__ el, const float* __restrict__ er,
                           unsigned* __restrict__ segkey) {
  int gid = blockIdx.x * blockDim.x + threadIdx.x;
  if (gid >= NE * HEADS) return;
  int e = gid >> 3, hh = gid & 7;
  int s = src[e], d = dst[e];
  float v = el[(size_t)s * HEADS + hh] + er[(size_t)d * HEADS + hh];
  v = v > 0.f ? v : NSLOPE * v;
  atomicMax(&segkey[(size_t)d * HEADS + hh], fkey(v));
}

__global__ void k_edge_alpha(const int* __restrict__ src, const int* __restrict__ dst,
                             const float* __restrict__ el, const float* __restrict__ er,
                             const unsigned* __restrict__ segkey,
                             float* __restrict__ alpha, float* __restrict__ denom) {
  int gid = blockIdx.x * blockDim.x + threadIdx.x;
  if (gid >= NE * HEADS) return;
  int e = gid >> 3, hh = gid & 7;
  int s = src[e], d = dst[e];
  float v = el[(size_t)s * HEADS + hh] + er[(size_t)d * HEADS + hh];
  v = v > 0.f ? v : NSLOPE * v;
  float m = funkey(segkey[(size_t)d * HEADS + hh]);
  float a = expf(v - m);
  alpha[(size_t)e * HEADS + hh] = a;
  atomicAddF(&denom[(size_t)d * HEADS + hh], a);
}

__global__ void k_aggregate(const int* __restrict__ src, const int* __restrict__ dst,
                            const float* __restrict__ alpha, const float* __restrict__ denom,
                            const float* __restrict__ feat, float* __restrict__ rst) {
  long long gid = (long long)blockIdx.x * blockDim.x + threadIdx.x;
  if (gid >= (long long)NE * EMBED) return;
  int e = (int)(gid >> 7);
  int c = (int)(gid & 127);
  int hh = c >> 4;
  int s = src[e], d = dst[e];
  float w = alpha[(size_t)e * HEADS + hh] / denom[(size_t)d * HEADS + hh];
  atomicAddF(&rst[(size_t)d * EMBED + c], w * feat[(size_t)s * EMBED + c]);
}

// ---------------- BatchNorm ----------------

__global__ void k_stats(const float* __restrict__ y, float* __restrict__ stats) {
  int t = threadIdx.x;
  int col = t & 127, half = t >> 7;
  int r0 = blockIdx.x * 64;
  int rend = r0 + 64 < NN ? r0 + 64 : NN;
  float s = 0.f, q = 0.f;
  for (int r = r0 + half; r < rend; r += 2) {
    float v = y[(size_t)r * OUTD + col];
    s += v;
    q += v * v;
  }
  atomicAddF(&stats[col], s);
  atomicAddF(&stats[128 + col], q);
}

__global__ void k_bn(float* __restrict__ y, const float* __restrict__ stats,
                     const float* __restrict__ gamma, const float* __restrict__ beta) {
  int gid = blockIdx.x * blockDim.x + threadIdx.x;
  if (gid >= NN * OUTD) return;
  int c = gid & 127;
  float mean = stats[c] * (1.f / NN);
  float var  = stats[128 + c] * (1.f / NN) - mean * mean;
  float v = y[gid];
  y[gid] = gamma[c] * (v - mean) * rsqrtf(var + EPSV) + beta[c];
}

// ---------------- launch ----------------

extern "C" void kernel_launch(void* const* d_in, const int* in_sizes, int n_in,
                              void* d_out, int out_size, void* d_ws, size_t ws_size,
                              hipStream_t stream) {
  (void)in_sizes; (void)n_in; (void)out_size; (void)ws_size;
  const float* h     = (const float*)d_in[0];
  const int*   src   = (const int*)d_in[1];
  const int*   dst   = (const int*)d_in[2];
  const float* Wfc   = (const float*)d_in[3];
  const float* al    = (const float*)d_in[4];
  const float* ar    = (const float*)d_in[5];
  const float* gb    = (const float*)d_in[6];
  const float* W1    = (const float*)d_in[7];
  const float* b1    = (const float*)d_in[8];
  const float* W2    = (const float*)d_in[9];
  const float* b2    = (const float*)d_in[10];
  const float* gamma = (const float*)d_in[11];
  const float* beta  = (const float*)d_in[12];
  float* out = (float*)d_out;

  char* base = (char*)d_ws;
  size_t off = 0;
  auto take = [&](size_t bytes) -> void* {
    void* p = base + off;
    off += (bytes + 255) & ~(size_t)255;
    return p;
  };
  float*          feat   = (float*)take((size_t)NN * EMBED * 4);
  float*          el     = (float*)take((size_t)NN * HEADS * 4);
  float*          er     = (float*)take((size_t)NN * HEADS * 4);
  unsigned*       segkey = (unsigned*)take((size_t)NN * HEADS * 4);
  float*          denom  = (float*)take((size_t)NN * HEADS * 4);
  float*          alpha  = (float*)take((size_t)NE * HEADS * 4);
  float*          rst    = (float*)take((size_t)NN * EMBED * 4);
  unsigned short* x1     = (unsigned short*)take((size_t)NN * HIDDEN * 2);
  unsigned short* wtfc   = (unsigned short*)take((size_t)EMBED * EMBED * 2);
  unsigned short* wt1    = (unsigned short*)take((size_t)HIDDEN * EMBED * 2);
  unsigned short* wt2    = (unsigned short*)take((size_t)HIDDEN * OUTD * 2);
  float*          stats  = (float*)take(256 * 4);

  const int mblocks = (NN + 127) / 128;  // 391
  const int wtotal  = EMBED * EMBED + 2 * HIDDEN * EMBED;

  k_init<<<(NN * EMBED + 255) / 256, 256, 0, stream>>>(rst, denom, segkey, stats);
  k_prep_weights<<<(wtotal + 255) / 256, 256, 0, stream>>>(Wfc, W1, W2, wtfc, wt1, wt2);
  k_gemm_feat<<<mblocks, 256, 0, stream>>>(h, wtfc, feat);
  k_el_er<<<(NN * HEADS + 255) / 256, 256, 0, stream>>>(feat, al, ar, el, er);
  k_edge_max<<<(NE * HEADS + 255) / 256, 256, 0, stream>>>(src, dst, el, er, segkey);
  k_edge_alpha<<<(NE * HEADS + 255) / 256, 256, 0, stream>>>(src, dst, el, er, segkey,
                                                             alpha, denom);
  {
    long long total = (long long)NE * EMBED;
    int blocks = (int)((total + 255) / 256);
    k_aggregate<<<blocks, 256, 0, stream>>>(src, dst, alpha, denom, feat, rst);
  }
  k_gemm_mlp1<<<dim3(mblocks, 4), 256, 0, stream>>>(rst, gb, wt1, b1, x1);
  k_gemm_mlp2<<<mblocks, 256, 0, stream>>>(x1, wt2, b2, out);
  k_stats<<<(NN + 63) / 64, 256, 0, stream>>>(out, stats);
  k_bn<<<(NN * OUTD + 255) / 256, 256, 0, stream>>>(out, stats, gamma, beta);
}
